// FlaxMamba2Mixer_58342835749419
// MI455X (gfx1250) — matrix operand order repaired and checked by reading, not compiled, not run on hardware
//
#include <hip/hip_runtime.h>
#include <hip/hip_bf16.h>

// ---------------- problem constants ----------------
#define B_      2
#define L_      2048
#define TOK     4096        // B_*L_
#define H_      2048
#define NH_     64
#define HD_     64
#define NS      128         // state dim N
#define INTER_  4096
#define CONVD   4352        // INTER + 2*G*N
#define PROJ_   8512        // INTER + CONVD + NH
#define CHUNK_  256
#define NCH     8           // chunks per batch
#define BC      16          // B_*NCH
#define EPS_    1e-5f

typedef unsigned short u16;
typedef __attribute__((ext_vector_type(16))) __bf16       v16bf;
typedef __attribute__((ext_vector_type(8)))  float        v8f;
typedef __attribute__((ext_vector_type(4)))  unsigned int u32x4;

union FragH { v16bf v; u16 u[16]; u32x4 q[2]; };

__device__ __forceinline__ u16 f2bf(float f) {
    unsigned int u = __float_as_uint(f);
    u += 0x7FFFu + ((u >> 16) & 1u);          // round-to-nearest-even
    return (u16)(u >> 16);
}
__device__ __forceinline__ float bf2f(u16 h) {
    return __uint_as_float(((unsigned int)h) << 16);
}
__device__ __forceinline__ v8f wmma_bf16(v16bf a, v16bf b, v8f c) {
    // (neg_a, A, neg_b, B, c_mod, C, reuse_a, reuse_b)
    return __builtin_amdgcn_wmma_f32_16x16x32_bf16(false, a, false, b, (short)0, c, false, false);
}
__device__ __forceinline__ float siluf(float x) { return x / (1.f + __expf(-x)); }

// ---------------- elementwise prep ----------------
__global__ void cast_bf16_kernel(const float* __restrict__ in, u16* __restrict__ out, long n) {
    long i = (long)blockIdx.x * blockDim.x + threadIdx.x;
    if (i < n) out[i] = f2bf(in[i]);
}

// in: R x C row-major (f32)  ->  out: C x R (bf16)   (N-major "Bt" layout for GEMM)
__global__ void transpose_cast_kernel(const float* __restrict__ in, u16* __restrict__ out,
                                      int R, int C) {
    long i = (long)blockIdx.x * blockDim.x + threadIdx.x;
    if (i < (long)R * C) {
        int r = (int)(i / C), c = (int)(i % C);
        out[(long)c * R + r] = f2bf(in[i]);
    }
}

// ---------------- generic batched bf16 WMMA GEMM ----------------
// A: M x K row-major bf16.  Bt: N x K row-major bf16 (i.e. B stored N-major).
// C: M x N f32.  M%256==0, N%64==0, K%32==0.
// Block = 256 threads = 8 waves laid out 4(M) x 2(N); wave tile = 64x32
// -> 8 WMMAs per K-step against 6 fragment loads (high intensity, compute-bound).
__global__ void gemm_bf16_kernel(const u16* __restrict__ A, const u16* __restrict__ Bt,
                                 float* __restrict__ C, int M, int Nn, int Kd,
                                 long sA, long sB, long sC) {
    long zb = blockIdx.z;
    A  += zb * sA; Bt += zb * sB; C += zb * sC;
    int lane = threadIdx.x & 31, wave = threadIdx.x >> 5;
    int wm = wave & 3, wn = wave >> 2;                 // 4 x 2 wave grid
    int m0 = blockIdx.y * 256 + wm * 64;
    int n0 = blockIdx.x * 64  + wn * 32;
    int ml = lane & 15, hi = lane >> 4;
    v8f acc[4][2] = {};
    for (int kt = 0; kt < Kd; kt += 32) {
        FragH a[4], b[2];
#pragma unroll
        for (int i = 0; i < 4; ++i) {
            const u16* pa = A + (long)(m0 + i * 16 + ml) * Kd + kt + hi * 8;
            a[i].q[0] = *(const u32x4*)pa;
            a[i].q[1] = *(const u32x4*)(pa + 16);
        }
#pragma unroll
        for (int j = 0; j < 2; ++j) {
            const u16* pb = Bt + (long)(n0 + j * 16 + ml) * Kd + kt + hi * 8;
            b[j].q[0] = *(const u32x4*)pb;
            b[j].q[1] = *(const u32x4*)(pb + 16);
        }
#pragma unroll
        for (int i = 0; i < 4; ++i)
#pragma unroll
            for (int j = 0; j < 2; ++j)
                acc[i][j] = wmma_bf16(a[i].v, b[j].v, acc[i][j]);
    }
#pragma unroll
    for (int i = 0; i < 4; ++i)
#pragma unroll
        for (int j = 0; j < 2; ++j) {
            int row0 = m0 + i * 16 + hi * 8;
            int col  = n0 + j * 16 + ml;
#pragma unroll
            for (int r = 0; r < 8; ++r)
                C[(long)(row0 + r) * Nn + col] = acc[i][j][r];
        }
}

// ---------------- dt softplus ----------------
__global__ void dtsp_kernel(const float* __restrict__ proj, const float* __restrict__ dt_bias,
                            float* __restrict__ dtsp) {
    long i = (long)blockIdx.x * blockDim.x + threadIdx.x;
    if (i < (long)TOK * NH_) {
        int h = (int)(i % NH_); long tok = i / NH_;
        float x = proj[tok * PROJ_ + (INTER_ + CONVD) + h] + dt_bias[h];
        dtsp[i] = (x > 20.f) ? x : log1pf(__expf(x));
    }
}

// ---------------- per-chunk cumsum of A*dt ----------------
__global__ void acum_kernel(const float* __restrict__ dtsp, const float* __restrict__ A_log,
                            float* __restrict__ acum) {
    int i = blockIdx.x * blockDim.x + threadIdx.x;     // over BC*NH
    if (i < BC * NH_) {
        int bc = i / NH_, h = i % NH_;
        float A = -__expf(A_log[h]);
        long tok0 = (long)bc * CHUNK_;
        float a = 0.f;
        for (int s = 0; s < CHUNK_; ++s) {
            a += A * dtsp[(tok0 + s) * NH_ + h];
            acum[((long)bc * NH_ + h) * CHUNK_ + s] = a;
        }
    }
}

// ---------------- depthwise causal conv + SiLU + split/cast ----------------
__global__ void conv_kernel(const float* __restrict__ proj, const float* __restrict__ conv_w,
                            const float* __restrict__ conv_b, const float* __restrict__ dtsp,
                            float* __restrict__ xs, u16* __restrict__ xdtb,
                            u16* __restrict__ Bmb, u16* __restrict__ Cmb) {
    long i = (long)blockIdx.x * blockDim.x + threadIdx.x;
    if (i >= (long)TOK * CONVD) return;
    int c = (int)(i % CONVD); long tok = i / CONVD;
    int l = (int)(tok % L_);
    float acc = conv_b[c];
#pragma unroll
    for (int k = 0; k < 4; ++k) {
        int back = 3 - k;
        if (l - back >= 0)
            acc += conv_w[c * 4 + k] * proj[(tok - back) * PROJ_ + INTER_ + c];
    }
    float v = siluf(acc);
    if (c < INTER_) {
        xs[tok * INTER_ + c] = v;
        int h = c >> 6;
        xdtb[tok * INTER_ + c] = f2bf(v * dtsp[tok * NH_ + h]);
    } else if (c < INTER_ + NS) {
        Bmb[tok * NS + (c - INTER_)] = f2bf(v);
    } else {
        Cmb[tok * NS + (c - INTER_ - NS)] = f2bf(v);
    }
}

// ---------------- Y_diag: (mask(CB^T) .* L) @ (x*dt), per (b,chunk,head) ----------------
__global__ void ydiag_kernel(const float* __restrict__ CBt, const u16* __restrict__ xdtb,
                             const float* __restrict__ acum, float* __restrict__ Y) {
    __shared__ u16  sx[CHUNK_ * HD_];     // 32 KB: x*dt tile for this head
    __shared__ float sac[CHUNK_];
    int bch = blockIdx.x;
    int h = bch & 63, bc = bch >> 6;
    long tok0 = (long)bc * CHUNK_;
    const float* cb = CBt + (long)bc * CHUNK_ * CHUNK_;
    for (int idx = threadIdx.x; idx < CHUNK_ * HD_; idx += blockDim.x) {
        int s = idx >> 6, d = idx & 63;
        sx[idx] = xdtb[(tok0 + s) * INTER_ + h * 64 + d];
    }
    if (threadIdx.x < CHUNK_)
        sac[threadIdx.x] = acum[((long)bc * NH_ + h) * CHUNK_ + threadIdx.x];
    __syncthreads();

    int lane = threadIdx.x & 31, wave = threadIdx.x >> 5;
    int ml = lane & 15, hi = lane >> 4;
    int m0 = wave * 32;                                  // 8 waves cover 256 rows
    v8f acc[2][4] = {};
    for (int kt = 0; kt < CHUNK_; kt += 32) {
        FragH a[2];
#pragma unroll
        for (int i = 0; i < 2; ++i) {
            int lrow = m0 + i * 16 + ml;
            float al = sac[lrow];
            const float* row = cb + (long)lrow * CHUNK_ + kt;
#pragma unroll
            for (int g = 0; g < 2; ++g)
#pragma unroll
                for (int j = 0; j < 8; ++j) {
                    int sl = g * 16 + hi * 8 + j;
                    int s  = kt + sl;
                    float v = (s <= lrow) ? row[sl] * __expf(al - sac[s]) : 0.f;
                    a[i].u[g * 8 + j] = f2bf(v);
                }
        }
        FragH b[4];
#pragma unroll
        for (int j = 0; j < 4; ++j) {
            int d = j * 16 + ml;
#pragma unroll
            for (int t = 0; t < 16; ++t)
                b[j].u[t] = sx[(kt + (t >> 3) * 16 + hi * 8 + (t & 7)) * 64 + d];
        }
#pragma unroll
        for (int i = 0; i < 2; ++i)
#pragma unroll
            for (int j = 0; j < 4; ++j)
                acc[i][j] = wmma_bf16(a[i].v, b[j].v, acc[i][j]);
    }
#pragma unroll
    for (int i = 0; i < 2; ++i)
#pragma unroll
        for (int j = 0; j < 4; ++j) {
            int row0 = m0 + i * 16 + hi * 8;
            int col  = j * 16 + ml;
#pragma unroll
            for (int r = 0; r < 8; ++r)
                Y[(tok0 + row0 + r) * INTER_ + h * 64 + col] = acc[i][j][r];
        }
}

// ---------------- chunk states: (x*dt*decay)^T @ B, per (b,chunk,head) ----------------
__global__ void states_kernel(const u16* __restrict__ xdtb, const u16* __restrict__ Bmb,
                              const float* __restrict__ acum, float* __restrict__ statesf) {
    __shared__ u16  sx[CHUNK_ * HD_];     // 32 KB
    __shared__ u16  sBm[CHUNK_ * NS];     // 64 KB
    __shared__ float sac[CHUNK_];
    int bch = blockIdx.x;
    int h = bch & 63, bc = bch >> 6;
    long tok0 = (long)bc * CHUNK_;
    for (int idx = threadIdx.x; idx < CHUNK_ * HD_; idx += blockDim.x) {
        int s = idx >> 6, d = idx & 63;
        sx[idx] = xdtb[(tok0 + s) * INTER_ + h * 64 + d];
    }
    for (int idx = threadIdx.x; idx < CHUNK_ * NS; idx += blockDim.x)
        sBm[idx] = Bmb[tok0 * NS + idx];              // chunk is contiguous
    if (threadIdx.x < CHUNK_)
        sac[threadIdx.x] = acum[((long)bc * NH_ + h) * CHUNK_ + threadIdx.x];
    __syncthreads();

    int lane = threadIdx.x & 31, wave = threadIdx.x >> 5;
    int ml = lane & 15, hi = lane >> 4;
    int m0 = (wave & 1) * 32;      // d rows (64)
    int n0 = (wave >> 1) * 32;     // n cols (128)
    float alast = sac[CHUNK_ - 1];
    v8f acc[2][2] = {};
    for (int kt = 0; kt < CHUNK_; kt += 32) {
        FragH a[2], b[2];
#pragma unroll
        for (int i = 0; i < 2; ++i) {
            int d = m0 + i * 16 + ml;
#pragma unroll
            for (int g = 0; g < 2; ++g)
#pragma unroll
                for (int j = 0; j < 8; ++j) {
                    int s = kt + g * 16 + hi * 8 + j;
                    float v = bf2f(sx[s * 64 + d]) * __expf(alast - sac[s]);
                    a[i].u[g * 8 + j] = f2bf(v);
                }
            int n = n0 + i * 16 + ml;
#pragma unroll
            for (int t = 0; t < 16; ++t)
                b[i].u[t] = sBm[(kt + (t >> 3) * 16 + hi * 8 + (t & 7)) * NS + n];
        }
#pragma unroll
        for (int i = 0; i < 2; ++i)
#pragma unroll
            for (int j = 0; j < 2; ++j)
                acc[i][j] = wmma_bf16(a[i].v, b[j].v, acc[i][j]);
    }
    float* out = statesf + (long)bch * HD_ * NS;
#pragma unroll
    for (int i = 0; i < 2; ++i)
#pragma unroll
        for (int j = 0; j < 2; ++j) {
            int row0 = m0 + i * 16 + hi * 8;
            int col  = n0 + j * 16 + ml;
#pragma unroll
            for (int r = 0; r < 8; ++r)
                out[(long)(row0 + r) * NS + col] = acc[i][j][r];
        }
}

// ---------------- sequential inter-chunk state scan (8 steps) ----------------
__global__ void scan_kernel(const float* __restrict__ statesf, const float* __restrict__ acum,
                            u16* __restrict__ stateinb) {
    long i = (long)blockIdx.x * blockDim.x + threadIdx.x;   // B*NH*HD*NS = 2^20
    if (i >= (long)B_ * NH_ * HD_ * NS) return;
    int n = (int)(i & 127);
    int d = (int)((i >> 7) & 63);
    int h = (int)((i >> 13) & 63);
    int b = (int)(i >> 19);
    float S = 0.f;
    for (int c = 0; c < NCH; ++c) {
        int bc = b * NCH + c;
        long off = (((long)bc * NH_ + h) * HD_ + d) * NS + n;
        stateinb[off] = f2bf(S);                            // state entering chunk c
        float dec = __expf(acum[((long)bc * NH_ + h) * CHUNK_ + (CHUNK_ - 1)]);
        S = S * dec + statesf[off];
    }
}

// ---------------- Y_off = exp(acum) .* (C @ S_in^T)  (+ D residual), per (b,chunk,head) ----
__global__ void yoff_kernel(const u16* __restrict__ Cmb, const u16* __restrict__ stateinb,
                            const float* __restrict__ acum, const float* __restrict__ xs,
                            const float* __restrict__ Dp, float* __restrict__ Y) {
    __shared__ float sac[CHUNK_];
    int bch = blockIdx.x;
    int h = bch & 63, bc = bch >> 6;
    long tok0 = (long)bc * CHUNK_;
    if (threadIdx.x < CHUNK_)
        sac[threadIdx.x] = acum[((long)bc * NH_ + h) * CHUNK_ + threadIdx.x];
    __syncthreads();
    float Dh = Dp[h];

    int lane = threadIdx.x & 31, wave = threadIdx.x >> 5;
    int ml = lane & 15, hi = lane >> 4;
    int m0 = wave * 32;
    const u16* Sb = stateinb + (long)bch * HD_ * NS;        // [d][n] = N-major for B operand
    v8f acc[2][4] = {};
    for (int kt = 0; kt < NS; kt += 32) {
        FragH a[2], b[4];
#pragma unroll
        for (int i = 0; i < 2; ++i) {
            const u16* pa = Cmb + (tok0 + m0 + i * 16 + ml) * NS + kt + hi * 8;
            a[i].q[0] = *(const u32x4*)pa;
            a[i].q[1] = *(const u32x4*)(pa + 16);
        }
#pragma unroll
        for (int j = 0; j < 4; ++j) {
            const u16* pb = Sb + (long)(j * 16 + ml) * NS + kt + hi * 8;
            b[j].q[0] = *(const u32x4*)pb;
            b[j].q[1] = *(const u32x4*)(pb + 16);
        }
#pragma unroll
        for (int i = 0; i < 2; ++i)
#pragma unroll
            for (int j = 0; j < 4; ++j)
                acc[i][j] = wmma_bf16(a[i].v, b[j].v, acc[i][j]);
    }
#pragma unroll
    for (int i = 0; i < 2; ++i)
#pragma unroll
        for (int j = 0; j < 4; ++j) {
            int row0 = m0 + i * 16 + hi * 8;
            int col  = j * 16 + ml;
#pragma unroll
            for (int r = 0; r < 8; ++r) {
                int row = row0 + r;
                long o = (tok0 + row) * INTER_ + h * 64 + col;
                Y[o] += __expf(sac[row]) * acc[i][j][r] + Dh * xs[o];
            }
        }
}

// ---------------- gate (SiLU) + RMSNorm + cast ----------------
__global__ void gatenorm_kernel(const float* __restrict__ Y, const float* __restrict__ proj,
                                const float* __restrict__ norm_w, u16* __restrict__ yb) {
    int tok = blockIdx.x;
    int tid = threadIdx.x;
    const float* yrow = Y + (long)tok * INTER_;
    const float* grow = proj + (long)tok * PROJ_;
    float v[16];
    float ss = 0.f;
#pragma unroll
    for (int j = 0; j < 16; ++j) {
        int i = tid + j * 256;
        float g = grow[i];
        float t = yrow[i] * siluf(g);
        v[j] = t; ss += t * t;
    }
#pragma unroll
    for (int o = 16; o > 0; o >>= 1) ss += __shfl_xor(ss, o, 32);
    __shared__ float red[8];
    int wave = tid >> 5, lane = tid & 31;
    if (lane == 0) red[wave] = ss;
    __syncthreads();
    float tot = 0.f;
#pragma unroll
    for (int w = 0; w < 8; ++w) tot += red[w];
    float rs = rsqrtf(tot * (1.f / INTER_) + EPS_);
    u16* orow = yb + (long)tok * INTER_;
#pragma unroll
    for (int j = 0; j < 16; ++j) {
        int i = tid + j * 256;
        orow[i] = f2bf(v[j] * rs * norm_w[i]);
    }
}

// ---------------- host-side orchestration ----------------
extern "C" void kernel_launch(void* const* d_in, const int* in_sizes, int n_in,
                              void* d_out, int out_size, void* d_ws, size_t ws_size,
                              hipStream_t stream) {
    const float* x       = (const float*)d_in[0];
    const float* in_w    = (const float*)d_in[1];
    const float* conv_w  = (const float*)d_in[2];
    const float* conv_b  = (const float*)d_in[3];
    const float* dt_bias = (const float*)d_in[4];
    const float* A_log   = (const float*)d_in[5];
    const float* Dp      = (const float*)d_in[6];
    const float* norm_w  = (const float*)d_in[7];
    const float* out_w   = (const float*)d_in[8];
    float* out = (float*)d_out;

    char* wsb = (char*)d_ws;
    size_t off = 0;
    auto carve = [&](size_t bytes) -> void* {
        void* p = wsb + off;
        off = (off + bytes + 255) & ~(size_t)255;
        return p;
    };
    u16*   xb       = (u16*)  carve((size_t)TOK * H_ * 2);
    u16*   wit      = (u16*)  carve((size_t)PROJ_ * H_ * 2);
    u16*   wot      = (u16*)  carve((size_t)H_ * INTER_ * 2);
    float* proj     = (float*)carve((size_t)TOK * PROJ_ * 4);
    float* dtsp     = (float*)carve((size_t)TOK * NH_ * 4);
    float* acum     = (float*)carve((size_t)BC * NH_ * CHUNK_ * 4);
    float* xs       = (float*)carve((size_t)TOK * INTER_ * 4);
    u16*   xdtb     = (u16*)  carve((size_t)TOK * INTER_ * 2);
    u16*   Bmb      = (u16*)  carve((size_t)TOK * NS * 2);
    u16*   Cmb      = (u16*)  carve((size_t)TOK * NS * 2);
    float* CBt      = (float*)carve((size_t)BC * CHUNK_ * CHUNK_ * 4);
    float* statesf  = (float*)carve((size_t)BC * NH_ * HD_ * NS * 4);
    u16*   stateinb = (u16*)  carve((size_t)BC * NH_ * HD_ * NS * 2);
    float* Yb       = (float*)carve((size_t)TOK * INTER_ * 4);
    u16*   yb       = (u16*)  carve((size_t)TOK * INTER_ * 2);

    const int T = 256;
    // 1. casts / transposes
    {
        long n = (long)TOK * H_;
        cast_bf16_kernel<<<dim3((unsigned)((n + T - 1) / T)), T, 0, stream>>>(x, xb, n);
        long ni = (long)H_ * PROJ_;
        transpose_cast_kernel<<<dim3((unsigned)((ni + T - 1) / T)), T, 0, stream>>>(in_w, wit, H_, PROJ_);
        long no = (long)INTER_ * H_;
        transpose_cast_kernel<<<dim3((unsigned)((no + T - 1) / T)), T, 0, stream>>>(out_w, wot, INTER_, H_);
    }
    // 2. in_proj GEMM: proj[TOK][PROJ]
    gemm_bf16_kernel<<<dim3(PROJ_ / 64, TOK / 256, 1), T, 0, stream>>>(
        xb, wit, proj, TOK, PROJ_, H_, 0, 0, 0);
    // 3. dt softplus, A cumsum
    {
        long n = (long)TOK * NH_;
        dtsp_kernel<<<dim3((unsigned)((n + T - 1) / T)), T, 0, stream>>>(proj, dt_bias, dtsp);
        acum_kernel<<<dim3((BC * NH_ + T - 1) / T), T, 0, stream>>>(dtsp, A_log, acum);
    }
    // 4. depthwise conv + SiLU + split (xs f32, x*dt / B / C bf16)
    {
        long n = (long)TOK * CONVD;
        conv_kernel<<<dim3((unsigned)((n + T - 1) / T)), T, 0, stream>>>(
            proj, conv_w, conv_b, dtsp, xs, xdtb, Bmb, Cmb);
    }
    // 5. CB^T per (b,chunk), shared across heads (G=1): batched GEMM, z=16
    gemm_bf16_kernel<<<dim3(CHUNK_ / 64, CHUNK_ / 256, BC), T, 0, stream>>>(
        Cmb, Bmb, CBt, CHUNK_, CHUNK_, NS,
        (long)CHUNK_ * NS, (long)CHUNK_ * NS, (long)CHUNK_ * CHUNK_);
    // 6. Y_diag
    ydiag_kernel<<<dim3(BC * NH_), T, 0, stream>>>(CBt, xdtb, acum, Yb);
    // 7. per-chunk states
    states_kernel<<<dim3(BC * NH_), T, 0, stream>>>(xdtb, Bmb, acum, statesf);
    // 8. inter-chunk scan -> bf16 incoming states
    {
        long n = (long)B_ * NH_ * HD_ * NS;
        scan_kernel<<<dim3((unsigned)((n + T - 1) / T)), T, 0, stream>>>(statesf, acum, stateinb);
    }
    // 9. Y_off + D residual
    yoff_kernel<<<dim3(BC * NH_), T, 0, stream>>>(Cmb, stateinb, acum, xs, Dp, Yb);
    // 10. gate + RMSNorm -> bf16
    gatenorm_kernel<<<dim3(TOK), T, 0, stream>>>(Yb, proj, norm_w, yb);
    // 11. out_proj GEMM -> d_out
    gemm_bf16_kernel<<<dim3(H_ / 64, TOK / 256, 1), T, 0, stream>>>(
        yb, wot, out, TOK, H_, INTER_, 0, 0, 0);
}
